// NodeModel_48636209660178
// MI455X (gfx1250) — matrix operand order repaired and checked by reading, not compiled, not run on hardware
//
#include <hip/hip_runtime.h>

#define N_NODES 50000
#define N_EDGES 800000

typedef __attribute__((ext_vector_type(16))) __bf16 v16bf;
typedef __attribute__((ext_vector_type(8)))  float  v8f;

union Frag { unsigned u[8]; v16bf v; };

// f32 -> bf16 via hardware converts (RNE); clang lowers these casts to
// v_cvt_bf16_f32 / v_cvt_pk_bf16_f32 on gfx1250.
__device__ __forceinline__ unsigned short f2bf(float f) {
  union { __bf16 h; unsigned short u; } p;
  p.h = (__bf16)f;
  return p.u;
}
__device__ __forceinline__ unsigned pk2bf(float lo, float hi) {
  union { __bf16 h[2]; unsigned u; } p;
  p.h[0] = (__bf16)lo;
  p.h[1] = (__bf16)hi;
  return p.u;
}

__device__ __forceinline__ v8f wmma_bf16(v16bf a, v16bf b, v8f c) {
  // (neg_a, A, neg_b, B, c_mod, C, reuse_a, reuse_b)
  return __builtin_amdgcn_wmma_f32_16x16x32_bf16(false, a, false, b, (short)0, c,
                                                 false, false);
}

// B fragment pre-packed in ws: dword index = ((c*NT + t)*32 + lane)*8 + v
__device__ __forceinline__ v16bf load_bfrag(const unsigned* __restrict__ p) {
  Frag f;
  const uint4* q = (const uint4*)p;
  uint4 a = q[0], b = q[1];
  f.u[0] = a.x; f.u[1] = a.y; f.u[2] = a.z; f.u[3] = a.w;
  f.u[4] = b.x; f.u[5] = b.y; f.u[6] = b.z; f.u[7] = b.w;
  return f.v;
}

__global__ __launch_bounds__(256) void zero_kernel(float* __restrict__ p, int n) {
  int i = blockIdx.x * 256 + threadIdx.x;
  if (i < n) p[i] = 0.0f;
}

// Pack one f32 weight matrix [K x N] row-major into bf16 B-fragments.
// For packed dword idx: v=idx&7, lane=(idx>>3)&31, t=(idx>>8)%NT, c=(idx>>8)/NT
// k = c*32 + (lane>=16 ? 16 : 0) + 2v ; n = t*16 + (lane&15); low half = even K.
__device__ __forceinline__ void pack_one(const float* __restrict__ W, int N, int NT,
                                         unsigned* __restrict__ dst, int idx) {
  int v = idx & 7;
  int l = (idx >> 3) & 31;
  int ct = idx >> 8;
  int t = ct % NT;
  int c = ct / NT;
  int k = c * 32 + ((l & 16) ? 16 : 0) + 2 * v;
  int n = t * 16 + (l & 15);
  dst[idx] = pk2bf(W[k * N + n], W[(k + 1) * N + n]);
}

__global__ __launch_bounds__(256) void pack_kernel(const float* __restrict__ W1,
                                                   const float* __restrict__ W2,
                                                   const float* __restrict__ W3,
                                                   const float* __restrict__ W4,
                                                   unsigned* __restrict__ wp) {
  int i = blockIdx.x * 256 + threadIdx.x;
  if (i < 6144)        pack_one(W1, 128, 8, wp,          i);          // 96x128: 3 kchunks, 8 ntiles
  else if (i < 10240)  pack_one(W2,  64, 4, wp + 6144,   i - 6144);   // 128x64: 4 kchunks, 4 ntiles
  else if (i < 18432)  pack_one(W3, 128, 8, wp + 10240,  i - 10240);  // 128x128: 4 kchunks, 8 ntiles
  else if (i < 22528)  pack_one(W4,  64, 4, wp + 18432,  i - 18432);  // 128x64
}

// A-fragment K offset within a 32-wide chunk for vgpr v and lane half (khalf=0/8)
#define AKOFF(v, khalf) (2 * (khalf) + 2 * (v))

// LDS row stride (bf16 units) for the 16x128 per-wave transpose buffer
#define LSTR 136

__global__ __launch_bounds__(256) void edge_kernel(
    const float* __restrict__ x, const int* __restrict__ ei,
    const float* __restrict__ ea, const float* __restrict__ b1,
    const float* __restrict__ b2, const unsigned* __restrict__ w1p,
    const unsigned* __restrict__ w2p, float* __restrict__ sum,
    float* __restrict__ cnt) {
  __shared__ __align__(16) unsigned short lds[8 * 16 * LSTR];
  const int lane = threadIdx.x & 31;
  const int wave = threadIdx.x >> 5;
  const int tile = blockIdx.x * 8 + wave;   // grid sized exactly: always valid
  const int tb = tile * 16;
  const int m = lane & 15;
  const int half = lane >> 4;
  const int khalf = half * 8;
  const int e = tb + m;
  const int xrow = ei[e];                    // edge_index[0][e]
  const int crow = ei[N_EDGES + e];          // edge_index[1][e]
  if (half == 0) unsafeAtomicAdd(&cnt[crow], 1.0f);

  // ---- layer 1: relu( [16x96]A @ [96x128]W1 + b1 ) ----
  v8f acc[8];
#pragma unroll
  for (int t = 0; t < 8; ++t) {
    float bv = b1[t * 16 + m];
#pragma unroll
    for (int i = 0; i < 8; ++i) acc[t][i] = bv;
  }
#pragma unroll
  for (int c = 0; c < 3; ++c) {
    Frag a;
    const float* src = (c < 2) ? (x + (size_t)xrow * 64 + c * 32)
                               : (ea + (size_t)e * 32);
#pragma unroll
    for (int v = 0; v < 8; ++v) {
      float2 f = *(const float2*)(src + AKOFF(v, khalf));
      a.u[v] = pk2bf(f.x, f.y);
    }
#pragma unroll
    for (int t = 0; t < 8; ++t) {
      v16bf b = load_bfrag(w1p + (((c * 8 + t) * 32 + lane) << 3));
      acc[t] = wmma_bf16(a.v, b, acc[t]);
    }
  }

  // relu + C->A transpose through per-wave LDS tile (row-major [16][LSTR] bf16)
  unsigned short* L = lds + wave * (16 * LSTR);
#pragma unroll
  for (int t = 0; t < 8; ++t)
#pragma unroll
    for (int v = 0; v < 8; ++v) {
      float val = fmaxf(acc[t][v], 0.0f);
      L[(v + khalf) * LSTR + t * 16 + m] = f2bf(val);   // row M=v+8*half, col N
    }

  // ---- layer 2: [16x128]h @ [128x64]W2 + b2 ----
  v8f acc2[4];
#pragma unroll
  for (int t = 0; t < 4; ++t) {
    float bv = b2[t * 16 + m];
#pragma unroll
    for (int i = 0; i < 8; ++i) acc2[t][i] = bv;
  }
#pragma unroll
  for (int c = 0; c < 4; ++c) {
    Frag a;
#pragma unroll
    for (int v = 0; v < 8; ++v) {
      int k = c * 32 + AKOFF(v, khalf);
      a.u[v] = *(const unsigned*)(L + m * LSTR + k);    // two consecutive bf16
    }
#pragma unroll
    for (int t = 0; t < 4; ++t) {
      v16bf b = load_bfrag(w2p + (((c * 4 + t) * 32 + lane) << 3));
      acc2[t] = wmma_bf16(a.v, b, acc2[t]);
    }
  }

  // scatter-add messages into per-node sum (L2-resident, native f32 atomics)
  int cols[8];
#pragma unroll
  for (int v = 0; v < 8; ++v) cols[v] = ei[N_EDGES + tb + khalf + v];
#pragma unroll
  for (int t = 0; t < 4; ++t)
#pragma unroll
    for (int v = 0; v < 8; ++v)
      unsafeAtomicAdd(&sum[(size_t)cols[v] * 64 + t * 16 + m], acc2[t][v]);
}

__global__ __launch_bounds__(256) void node_kernel(
    const float* __restrict__ x, const float* __restrict__ sum,
    const float* __restrict__ cnt, const float* __restrict__ b3,
    const float* __restrict__ b4, const unsigned* __restrict__ w3p,
    const unsigned* __restrict__ w4p, float* __restrict__ out) {
  __shared__ __align__(16) unsigned short lds[8 * 16 * LSTR];
  const int lane = threadIdx.x & 31;
  const int wave = threadIdx.x >> 5;
  const int tile = blockIdx.x * 8 + wave;
  if (tile >= N_NODES / 16) return;          // wave-uniform exit, EXEC stays full
  const int tb = tile * 16;
  const int m = lane & 15;
  const int half = lane >> 4;
  const int khalf = half * 8;
  const int node = tb + m;
  const float scale = 1.0f / fmaxf(cnt[node], 1.0f);

  // ---- layer 1: relu( [16x128]concat(x,agg) @ [128x128]W3 + b3 ) ----
  v8f acc[8];
#pragma unroll
  for (int t = 0; t < 8; ++t) {
    float bv = b3[t * 16 + m];
#pragma unroll
    for (int i = 0; i < 8; ++i) acc[t][i] = bv;
  }
#pragma unroll
  for (int c = 0; c < 4; ++c) {
    Frag a;
    if (c < 2) {
      const float* src = x + (size_t)node * 64 + c * 32;
#pragma unroll
      for (int v = 0; v < 8; ++v) {
        float2 f = *(const float2*)(src + AKOFF(v, khalf));
        a.u[v] = pk2bf(f.x, f.y);
      }
    } else {
      const float* src = sum + (size_t)node * 64 + (c - 2) * 32;
#pragma unroll
      for (int v = 0; v < 8; ++v) {
        float2 f = *(const float2*)(src + AKOFF(v, khalf));
        a.u[v] = pk2bf(f.x * scale, f.y * scale);       // agg = sum / max(cnt,1)
      }
    }
#pragma unroll
    for (int t = 0; t < 8; ++t) {
      v16bf b = load_bfrag(w3p + (((c * 8 + t) * 32 + lane) << 3));
      acc[t] = wmma_bf16(a.v, b, acc[t]);
    }
  }

  unsigned short* L = lds + wave * (16 * LSTR);
#pragma unroll
  for (int t = 0; t < 8; ++t)
#pragma unroll
    for (int v = 0; v < 8; ++v) {
      float val = fmaxf(acc[t][v], 0.0f);
      L[(v + khalf) * LSTR + t * 16 + m] = f2bf(val);
    }

  // ---- layer 2: [16x128]g @ [128x64]W4 + b4 ----
  v8f acc2[4];
#pragma unroll
  for (int t = 0; t < 4; ++t) {
    float bv = b4[t * 16 + m];
#pragma unroll
    for (int i = 0; i < 8; ++i) acc2[t][i] = bv;
  }
#pragma unroll
  for (int c = 0; c < 4; ++c) {
    Frag a;
#pragma unroll
    for (int v = 0; v < 8; ++v) {
      int k = c * 32 + AKOFF(v, khalf);
      a.u[v] = *(const unsigned*)(L + m * LSTR + k);
    }
#pragma unroll
    for (int t = 0; t < 4; ++t) {
      v16bf b = load_bfrag(w4p + (((c * 4 + t) * 32 + lane) << 3));
      acc2[t] = wmma_bf16(a.v, b, acc2[t]);
    }
  }

  // write f32 output: row M = tb + khalf + v, col N = t*16 + m
#pragma unroll
  for (int t = 0; t < 4; ++t)
#pragma unroll
    for (int v = 0; v < 8; ++v)
      out[(size_t)(tb + khalf + v) * 64 + t * 16 + m] = acc2[t][v];
}

extern "C" void kernel_launch(void* const* d_in, const int* in_sizes, int n_in,
                              void* d_out, int out_size, void* d_ws, size_t ws_size,
                              hipStream_t stream) {
  const float* x  = (const float*)d_in[0];
  const int*   ei = (const int*)d_in[1];
  const float* ea = (const float*)d_in[2];
  const float* W1 = (const float*)d_in[3];
  const float* b1 = (const float*)d_in[4];
  const float* W2 = (const float*)d_in[5];
  const float* b2 = (const float*)d_in[6];
  const float* W3 = (const float*)d_in[7];
  const float* b3 = (const float*)d_in[8];
  const float* W4 = (const float*)d_in[9];
  const float* b4 = (const float*)d_in[10];
  float* out = (float*)d_out;
  float* ws  = (float*)d_ws;

  // workspace layout (dwords):
  float*    sumbuf = ws;                                   // N*64 f32 scatter sums
  float*    cntbuf = ws + (size_t)N_NODES * 64;            // N f32 counts
  unsigned* wp     = (unsigned*)(ws + (size_t)N_NODES * 64 + N_NODES);
  unsigned* w1p = wp;            // 6144 dwords
  unsigned* w2p = wp + 6144;     // 4096
  unsigned* w3p = wp + 10240;    // 8192
  unsigned* w4p = wp + 18432;    // 4096

  const int zn = N_NODES * 65;   // sums + counts, contiguous
  zero_kernel<<<(zn + 255) / 256, 256, 0, stream>>>(sumbuf, zn);
  pack_kernel<<<(22528 + 255) / 256, 256, 0, stream>>>(W1, W2, W3, W4, wp);
  edge_kernel<<<N_EDGES / 128, 256, 0, stream>>>(x, ei, ea, b1, b2, w1p, w2p,
                                                 sumbuf, cntbuf);
  node_kernel<<<(N_NODES / 16 + 7) / 8, 256, 0, stream>>>(x, sumbuf, cntbuf,
                                                          b3, b4, w3p, w4p, out);
}
